// PatchAttention_46076409151821
// MI455X (gfx1250) — compile-verified, matrix operand order checked by reading, not run on hardware
//
#include <hip/hip_runtime.h>
#include <hip/hip_bf16.h>

typedef __attribute__((ext_vector_type(16))) __bf16 v16bf;
typedef __attribute__((ext_vector_type(8)))  __bf16 v8bf;
typedef __attribute__((ext_vector_type(8)))  float  v8f;

namespace {
constexpr int PS   = 7;          // patch size
constexpr int TOK  = 49;         // tokens per patch
constexpr int TPAD = 64;         // padded token rows
constexpr int CH   = 256;        // channels
constexpr int IMG  = 224;
constexpr int HW   = IMG * IMG;

// LDS strides (elements)
constexpr int SB  = 264;         // bf16 token matrices [64][SB]
constexpr int SVT = 72;          // bf16 V-transposed [256][SVT]
constexpr int SF  = 260;         // f32 token matrices [64][SF] (16B-aligned rows)
constexpr int SS  = 65;          // f32 scores [64][SS]
constexpr int SAT = 72;          // bf16 attn [64][SAT]

// LDS byte offsets
constexpr int OFF_SEG = 0;                            // seg bf16 -> reused as X bf16
constexpr int OFF_GAU = OFF_SEG + TPAD * SB * 2;      // gauss bf16 -> reused as H bf16
constexpr int OFF_Q   = OFF_GAU + TPAD * SB * 2;      // Q bf16 -> reused as attn bf16
constexpr int OFF_K   = OFF_Q   + TPAD * SB * 2;      // K bf16
constexpr int OFF_VT  = OFF_K   + TPAD * SB * 2;      // V^T bf16 [256][SVT]
constexpr int OFF_V32 = OFF_VT  + CH * SVT * 2;       // V f32 -> reused as FFN-out f32
constexpr int OFF_X32 = OFF_V32 + TPAD * SF * 4;      // attn-out f32 -> X f32 -> final
constexpr int OFF_S32 = OFF_X32 + TPAD * SF * 4;      // scores f32 -> LN partial sums
constexpr int LDS_BYTES = OFF_S32 + TPAD * SS * 4;
static_assert(LDS_BYTES <= 320 * 1024, "exceeds 320KB WGP LDS");
} // namespace

// ---- fragment load: two contiguous-K 16B chunks per lane (ISA 7.12.2) ----
__device__ __forceinline__ v16bf load_frag(const __bf16* base, int stride,
                                           int r0, int k0, int lane) {
    const __bf16* p = base + (r0 + (lane & 15)) * stride + k0 + ((lane & 16) ? 8 : 0);
    v16bf f;
    *(v8bf*)&f         = *(const v8bf*)p;          // K-chunk lo
    *(((v8bf*)&f) + 1) = *(const v8bf*)(p + 16);   // K-chunk hi
    return f;
}

__device__ __forceinline__ v8f wmma_bf16(v16bf a, v16bf b, v8f c) {
    return __builtin_amdgcn_wmma_f32_16x16x32_bf16(false, a, false, b, (short)0, c,
                                                   false, false);
}

// ---- 64xNxK GEMM, per-wave 2 M-tiles x 4 N-tiles (B fragment feeds 2 WMMAs) ----
// A: LDS bf16 (strideA). B: bf16 row-major over N with contiguous K (global or LDS).
template <int KEXT, bool PREF, bool BIAS, bool RELU,
          bool STORE_BF, bool STORE_F32, bool STORE_T>
__device__ __forceinline__ void gemm2x4(const __bf16* sA, int strideA,
                                        const __bf16* B, int strideB,
                                        const float* bias, __bf16* oBF,
                                        float* oF32, __bf16* oT,
                                        int wv, int lane) {
    const int mtb = (wv & 1) * 2;        // m-tiles {0,1} or {2,3}
    const int ntb = (wv >> 1) * 4;       // n-tile group of 4 (covers 16)
    v8f acc[2][4];
#pragma unroll
    for (int m = 0; m < 2; ++m)
#pragma unroll
        for (int j = 0; j < 4; ++j)
            acc[m][j] = v8f{0.f, 0.f, 0.f, 0.f, 0.f, 0.f, 0.f, 0.f};

    for (int k0 = 0; k0 < KEXT; k0 += 32) {
        if (PREF)
            __builtin_prefetch(B + (ntb * 16 + (lane & 15)) * strideB + k0 + 64, 0, 1);
        v16bf a0 = load_frag(sA, strideA, (mtb + 0) * 16, k0, lane);
        v16bf a1 = load_frag(sA, strideA, (mtb + 1) * 16, k0, lane);
#pragma unroll
        for (int j = 0; j < 4; ++j) {
            v16bf bf = load_frag(B, strideB, (ntb + j) * 16, k0, lane);
            acc[0][j] = wmma_bf16(a0, bf, acc[0][j]);
            acc[1][j] = wmma_bf16(a1, bf, acc[1][j]);
        }
    }
#pragma unroll
    for (int m = 0; m < 2; ++m) {
#pragma unroll
        for (int j = 0; j < 4; ++j) {
            const int n  = (ntb + j) * 16 + (lane & 15);
            const int rb = (mtb + m) * 16 + ((lane & 16) ? 8 : 0);
            const float bv = BIAS ? bias[n] : 0.f;
            v8f a8 = acc[m][j];
#pragma unroll
            for (int i = 0; i < 8; ++i) {
                float v = a8[i] + bv;
                if (RELU) v = fmaxf(v, 0.f);
                a8[i] = v;
                if (STORE_F32) oF32[(rb + i) * SF + n] = v;
                if (STORE_BF)  oBF[(rb + i) * SB + n] = (__bf16)v;
            }
            if (STORE_T) {   // column-contiguous store: one 16B chunk per lane
                v8bf t8;
#pragma unroll
                for (int i = 0; i < 8; ++i) t8[i] = (__bf16)a8[i];
                *(v8bf*)(oT + n * SVT + rb) = t8;
            }
        }
    }
}

__global__ void cvt_f32_bf16(const float* __restrict__ s, __bf16* __restrict__ d, int n) {
    int i = blockIdx.x * blockDim.x + threadIdx.x;
    if (i < n) d[i] = (__bf16)s[i];
}

__global__ __launch_bounds__(256, 1)
void patch_attn_kernel(const float* __restrict__ seg, const float* __restrict__ gau,
                       const __bf16* __restrict__ qwb, const __bf16* __restrict__ kwb,
                       const __bf16* __restrict__ vwb, const __bf16* __restrict__ w1b,
                       const __bf16* __restrict__ w2b,
                       const float* __restrict__ qb, const float* __restrict__ kb,
                       const float* __restrict__ vb, const float* __restrict__ b1,
                       const float* __restrict__ b2,
                       const float* __restrict__ ln1w, const float* __restrict__ ln1b,
                       const float* __restrict__ ln2w, const float* __restrict__ ln2b,
                       float* __restrict__ out) {
    extern __shared__ __align__(16) char smem[];
    __bf16* sSeg = (__bf16*)(smem + OFF_SEG);
    __bf16* sGau = (__bf16*)(smem + OFF_GAU);
    __bf16* sQ   = (__bf16*)(smem + OFF_Q);
    __bf16* sK   = (__bf16*)(smem + OFF_K);
    __bf16* sVT  = (__bf16*)(smem + OFF_VT);
    float*  sV32 = (float*)(smem + OFF_V32);
    float*  sX32 = (float*)(smem + OFF_X32);
    float*  sS32 = (float*)(smem + OFF_S32);
    __bf16* sAttn = sQ;    // reuse after scores
    __bf16* sX    = sSeg;  // reuse after LN1
    __bf16* sH    = sGau;  // reuse for FFN hidden
    float*  sF32  = sV32;  // reuse for FFN output

    const int tid  = threadIdx.x;
    const int lane = tid & 31;
    const int wv   = tid >> 5;

    const int blk = blockIdx.x;
    const int b   = blk >> 10;
    const int ph  = (blk >> 5) & 31;
    const int pw  = blk & 31;
    const long base = (long)b * CH * HW + (long)(ph * PS) * IMG + pw * PS;
    const float* segB = seg + base;
    const float* gauB = gau + base;

    // ---- load patch tokens (fp32 -> bf16), zero pad rows 49..63 ----
    for (int idx = tid; idx < TPAD * CH; idx += 256) {
        const int t = idx & 63;
        const int c = idx >> 6;
        if (t < TOK) {
            const long g = (long)c * HW + (t / PS) * IMG + (t % PS);
            sSeg[t * SB + c] = (__bf16)segB[g];
            sGau[t * SB + c] = (__bf16)gauB[g];
        } else {
            sSeg[t * SB + c] = (__bf16)0.f;
            sGau[t * SB + c] = (__bf16)0.f;
        }
    }
    __syncthreads();

    // ---- Q, K, V projections ----
    gemm2x4<CH, true, true, false, true, false, false>(
        sSeg, SB, qwb, CH, qb, sQ, nullptr, nullptr, wv, lane);
    gemm2x4<CH, true, true, false, true, false, false>(
        sSeg, SB, kwb, CH, kb, sK, nullptr, nullptr, wv, lane);
    gemm2x4<CH, true, true, false, false, true, true>(
        sGau, SB, vwb, CH, vb, nullptr, sV32, sVT, wv, lane);
    __syncthreads();

    // ---- scores = Q K^T : 64x64, K-dim 256 ----
    {
        const int mt  = wv & 3;
        const int ntb = (wv >> 2) * 2;
        v8f acc[2];
#pragma unroll
        for (int j = 0; j < 2; ++j) acc[j] = v8f{0.f, 0.f, 0.f, 0.f, 0.f, 0.f, 0.f, 0.f};
        for (int k0 = 0; k0 < CH; k0 += 32) {
            v16bf a = load_frag(sQ, SB, mt * 16, k0, lane);
#pragma unroll
            for (int j = 0; j < 2; ++j) {
                v16bf bb = load_frag(sK, SB, (ntb + j) * 16, k0, lane);
                acc[j] = wmma_bf16(a, bb, acc[j]);
            }
        }
#pragma unroll
        for (int j = 0; j < 2; ++j) {
            const int n  = (ntb + j) * 16 + (lane & 15);
            const int rb = mt * 16 + ((lane & 16) ? 8 : 0);
#pragma unroll
            for (int i = 0; i < 8; ++i) sS32[(rb + i) * SS + n] = acc[j][i];
        }
    }
    __syncthreads();

    // ---- softmax over s<49, scale = 1/sqrt(256); zero padded cols/rows ----
    if (tid < TPAD) {
        const int t = tid;
        __bf16* arow = sAttn + t * SAT;
        if (t < TOK) {
            float* srow = sS32 + t * SS;
            const float scale = 0.0625f;
            float mx = -1e30f;
            for (int s = 0; s < TOK; ++s) mx = fmaxf(mx, srow[s]);
            mx *= scale;
            float sum = 0.f;
            for (int s = 0; s < TOK; ++s) {
                const float e = __expf(srow[s] * scale - mx);
                srow[s] = e;
                sum += e;
            }
            const float r = 1.f / sum;
            for (int s = 0; s < TOK; ++s) arow[s] = (__bf16)(srow[s] * r);
            for (int s = TOK; s < TPAD; ++s) arow[s] = (__bf16)0.f;
        } else {
            for (int s = 0; s < TPAD; ++s) arow[s] = (__bf16)0.f;
        }
    }
    __syncthreads();

    // ---- out = attn @ V : A = attn [64x64], B-operand = V^T (contiguous-K rows) ----
    gemm2x4<TPAD, false, false, false, false, true, false>(
        sAttn, SAT, sVT, SVT, nullptr, nullptr, sX32, nullptr, wv, lane);
    __syncthreads();

    // ---- LN1 (4 threads/row): x = LN(v + out) -> f32 residual + bf16 FFN operand ----
    {
        const int t = tid & 63;
        const int part = tid >> 6;      // 0..3, 64 channels each
        const float4* v4 = (const float4*)(sV32 + t * SF + part * 64);
        const float4* o4 = (const float4*)(sX32 + t * SF + part * 64);
        float sum = 0.f, sq = 0.f;
        for (int q = 0; q < 16; ++q) {
            const float4 a = v4[q], bq = o4[q];
            const float x0 = a.x + bq.x, x1 = a.y + bq.y;
            const float x2 = a.z + bq.z, x3 = a.w + bq.w;
            sum += x0 + x1 + x2 + x3;
            sq  += x0 * x0 + x1 * x1 + x2 * x2 + x3 * x3;
        }
        sS32[t * 8 + part]     = sum;
        sS32[t * 8 + 4 + part] = sq;
        __syncthreads();
        float s4 = 0.f, q4 = 0.f;
        for (int p = 0; p < 4; ++p) { s4 += sS32[t * 8 + p]; q4 += sS32[t * 8 + 4 + p]; }
        const float mu  = s4 * (1.f / CH);
        const float var = q4 * (1.f / CH) - mu * mu;
        const float rs  = rsqrtf(var + 1e-5f);
        float* orow = sX32 + t * SF;
        const float* vr = sV32 + t * SF;
        for (int c = part * 64; c < part * 64 + 64; ++c) {
            const float x = (vr[c] + orow[c] - mu) * rs * ln1w[c] + ln1b[c];
            orow[c] = x;
            sX[t * SB + c] = (__bf16)x;
        }
    }
    __syncthreads();

    // ---- FFN ----
    gemm2x4<CH, true, true, true, true, false, false>(
        sX, SB, w1b, CH, b1, sH, nullptr, nullptr, wv, lane);
    __syncthreads();
    gemm2x4<CH, true, true, false, false, true, false>(
        sH, SB, w2b, CH, b2, nullptr, sF32, nullptr, wv, lane);
    __syncthreads();

    // ---- LN2 (4 threads/row): final = LN(x + ffn), written in place into sX32 ----
    {
        const int t = tid & 63;
        const int part = tid >> 6;
        const float4* x4 = (const float4*)(sX32 + t * SF + part * 64);
        const float4* f4 = (const float4*)(sF32 + t * SF + part * 64);
        float sum = 0.f, sq = 0.f;
        for (int q = 0; q < 16; ++q) {
            const float4 a = x4[q], bq = f4[q];
            const float x0 = a.x + bq.x, x1 = a.y + bq.y;
            const float x2 = a.z + bq.z, x3 = a.w + bq.w;
            sum += x0 + x1 + x2 + x3;
            sq  += x0 * x0 + x1 * x1 + x2 * x2 + x3 * x3;
        }
        sS32[t * 8 + part]     = sum;
        sS32[t * 8 + 4 + part] = sq;
        __syncthreads();
        float s4 = 0.f, q4 = 0.f;
        for (int p = 0; p < 4; ++p) { s4 += sS32[t * 8 + p]; q4 += sS32[t * 8 + 4 + p]; }
        const float mu  = s4 * (1.f / CH);
        const float var = q4 * (1.f / CH) - mu * mu;
        const float rs  = rsqrtf(var + 1e-5f);
        float* xr = sX32 + t * SF;
        const float* fr = sF32 + t * SF;
        for (int c = part * 64; c < part * 64 + 64; ++c)
            xr[c] = (xr[c] + fr[c] - mu) * rs * ln2w[c] + ln2b[c];
    }
    __syncthreads();

    // ---- writeback to [B,C,H,W] ----
    float* outB = out + base;
    for (int idx = tid; idx < TOK * CH; idx += 256) {
        const int t = idx % TOK;
        const int c = idx / TOK;
        outB[(long)c * HW + (t / PS) * IMG + (t % PS)] = sX32[t * SF + c];
    }
}

extern "C" void kernel_launch(void* const* d_in, const int* in_sizes, int n_in,
                              void* d_out, int out_size, void* d_ws, size_t ws_size,
                              hipStream_t stream) {
    const float* seg  = (const float*)d_in[0];
    const float* gau  = (const float*)d_in[1];
    const float* qw   = (const float*)d_in[2];
    const float* qb   = (const float*)d_in[3];
    const float* kw   = (const float*)d_in[4];
    const float* kb   = (const float*)d_in[5];
    const float* vw   = (const float*)d_in[6];
    const float* vb   = (const float*)d_in[7];
    const float* ln1w = (const float*)d_in[8];
    const float* ln1b = (const float*)d_in[9];
    const float* ln2w = (const float*)d_in[10];
    const float* ln2b = (const float*)d_in[11];
    const float* w1   = (const float*)d_in[12];
    const float* b1   = (const float*)d_in[13];
    const float* w2   = (const float*)d_in[14];
    const float* b2   = (const float*)d_in[15];
    float* out = (float*)d_out;

    const int WN = CH * CH; // 65536
    __bf16* ws  = (__bf16*)d_ws;
    __bf16* qwb = ws + 0 * WN;
    __bf16* kwb = ws + 1 * WN;
    __bf16* vwb = ws + 2 * WN;
    __bf16* w1b = ws + 3 * WN;
    __bf16* w2b = ws + 4 * WN;

    cvt_f32_bf16<<<WN / 256, 256, 0, stream>>>(qw, qwb, WN);
    cvt_f32_bf16<<<WN / 256, 256, 0, stream>>>(kw, kwb, WN);
    cvt_f32_bf16<<<WN / 256, 256, 0, stream>>>(vw, vwb, WN);
    cvt_f32_bf16<<<WN / 256, 256, 0, stream>>>(w1, w1b, WN);
    cvt_f32_bf16<<<WN / 256, 256, 0, stream>>>(w2, w2b, WN);

    const int nBlocks = 8 * 32 * 32; // B * nh * nw = 8192 patches
    patch_attn_kernel<<<nBlocks, 256, LDS_BYTES, stream>>>(
        seg, gau, qwb, kwb, vwb, w1b, w2b,
        qb, kb, vb, b1, b2, ln1w, ln1b, ln2w, ln2b, out);
}
